// Graphormer_2619930050607
// MI455X (gfx1250) — matrix operand order repaired and checked by reading, not compiled, not run on hardware
//
#include <hip/hip_runtime.h>

// ---------------- problem constants (match reference) ----------------
#define NN      50000
#define EE      800000
#define INC     128
#define HID     256
#define HEADS   4
#define HD      64
#define NLAYER  2
#define NGRAPH  64
#define NCLS    10
#define EA      (EE + NN)          // edges + self loops
#define NEG_SLOPE 0.2f

typedef __attribute__((ext_vector_type(16))) _Float16 v16h;
typedef __attribute__((ext_vector_type(8)))  _Float16 v8h;
typedef __attribute__((ext_vector_type(8)))  float    v8f;

// ---------------- device helpers ----------------
__device__ __forceinline__ void atomAddF(float* p, float v) {
  __hip_atomic_fetch_add(p, v, __ATOMIC_RELAXED, __HIP_MEMORY_SCOPE_AGENT);
}
// monotonic float -> uint encoding so atomicMax(uint) == float max
__device__ __forceinline__ unsigned fenc(float f) {
  int i = __float_as_int(f);
  return (unsigned)(i < 0 ? ~i : (i | 0x80000000));
}
__device__ __forceinline__ float fdec(unsigned u) {
  int i = (u & 0x80000000u) ? (int)(u & 0x7fffffffu) : ~(int)u;
  return __int_as_float(i);
}

// ---------------- f32 -> f16 cast ----------------
__global__ void k_cast_f16(const float* __restrict__ src, _Float16* __restrict__ dst, int n) {
  for (int i = blockIdx.x * blockDim.x + threadIdx.x; i < n; i += gridDim.x * blockDim.x)
    dst[i] = (_Float16)src[i];
}

// ---------------- pack weight [K,Nc] f32 into WMMA B-fragment layout ----------------
// packed index = ((kt*NT + nt)*32 + lane)*16 + j
// lane: col n = nt*16 + (lane&15);  k = kt*32 + (j>>3)*16 + (lane>>4)*8 + (j&7)
__global__ void k_pack_w(const float* __restrict__ W, _Float16* __restrict__ P, int K, int Nc) {
  int total = K * Nc;
  int NT = Nc >> 4;
  for (int idx = blockIdx.x * blockDim.x + threadIdx.x; idx < total; idx += gridDim.x * blockDim.x) {
    int j    = idx & 15;
    int lane = (idx >> 4) & 31;
    int rest = idx >> 9;
    int nt   = rest % NT;
    int kt   = rest / NT;
    int k = kt * 32 + (j >> 3) * 16 + (lane >> 4) * 8 + (j & 7);
    int n = nt * 16 + (lane & 15);
    P[idx] = (_Float16)W[(size_t)k * Nc + n];
  }
}

// ---------------- WMMA GEMM: C[M,Nc] = A[M,K](f16,row-major) @ Bpacked + epilogue --------
// One wave computes a 16x64 strip (4 x 16x16 tiles); A fragment shared across 4 WMMAs.
// KT (= K/32) is a compile-time constant so the k-loop fully unrolls: no register
// rotation copies, loads of later k-steps scheduled above earlier WMMAs.
template <int KT>
__global__ void __launch_bounds__(256)
k_wmma_gemm4(const _Float16* __restrict__ A, const _Float16* __restrict__ Bp,
             const float* __restrict__ bias, const float* __restrict__ degemb,
             const int* __restrict__ deg,
             float* __restrict__ C, _Float16* __restrict__ Cf16,
             int M, int Nc) {
  const int K     = KT * 32;
  const int NT    = Nc >> 4;            // 16-wide n tiles
  const int NT4   = Nc >> 6;            // 64-wide n strips
  const int strips = (M >> 4) * NT4;
  const int strip  = blockIdx.x * blockDim.y + threadIdx.y;  // wave-uniform
  if (strip >= strips) return;                                // whole wave exits together
  const int lane = threadIdx.x;
  const int mt = strip / NT4, ng = strip % NT4;
  const int m0 = mt << 4;
  const int ntb = ng << 2;              // first 16-wide n tile of this strip

  const int sel  = lane >> 4;           // which K-half of the 32-wide slab
  const int mrow = m0 + (lane & 15);    // A: lanes 0-15 / 16-31 both map rows 0-15
  const _Float16* arow = A + (size_t)mrow * K + sel * 8;
  const _Float16* brow = Bp + (((size_t)ntb * 32) + lane) * 16;

  v8f acc[4] = {};

#pragma unroll
  for (int kt = 0; kt < KT; ++kt) {
    // A fragment: halves 0..7 = K[kt*32 + sel*8 ..], halves 8..15 = +16
    v8h alo = *(const v8h*)(arow + kt * 32);
    v8h ahi = *(const v8h*)(arow + kt * 32 + 16);
    v16h av = __builtin_shufflevector(alo, ahi, 0,1,2,3,4,5,6,7,8,9,10,11,12,13,14,15);
    if (kt + 1 < KT)
      __builtin_prefetch((const void*)(arow + (kt + 1) * 32), 0, 3);
#pragma unroll
    for (int g = 0; g < 4; ++g) {
      // B fragment: 16 contiguous halves per lane from the packed buffer (32B load)
      v16h bv = *(const v16h*)(brow + ((size_t)kt * NT + g) * (32 * 16));
      acc[g] = __builtin_amdgcn_wmma_f32_16x16x32_f16(
          /*neg_a=*/false, av, /*neg_b=*/false, bv,
          /*c_mod=*/(short)0, acc[g], /*reuse_a=*/false, /*reuse_b=*/false);
    }
  }

  // C/D layout: VGPR r -> row m0 + (lane>>4)*8 + r ; col = tile_n0 + (lane&15)
  const int half = lane >> 4;
#pragma unroll
  for (int g = 0; g < 4; ++g) {
    const int ncol = ((ntb + g) << 4) + (lane & 15);
    float badd = bias ? bias[ncol] : 0.0f;
#pragma unroll
    for (int r = 0; r < 8; ++r) {
      int mr = m0 + half * 8 + r;
      float v = acc[g][r] + badd;
      if (degemb) v += degemb[(size_t)deg[mr] * Nc + ncol];
      C[(size_t)mr * Nc + ncol] = v;
      if (Cf16) Cf16[(size_t)mr * Nc + ncol] = (_Float16)v;
    }
  }
}

// ---------------- edge pass 1: logits + scatter max ----------------
// one wave per edge; lane handles 8 contiguous channels; head = lane>>3
__global__ void __launch_bounds__(256)
k_edge_logits(const float* __restrict__ XL, const float* __restrict__ XR,
              const int* __restrict__ ei, const float* __restrict__ att,
              float* __restrict__ logit, unsigned* __restrict__ menc) {
  int e = blockIdx.x * blockDim.y + threadIdx.y;
  if (e >= EA) return;
  int lane = threadIdx.x;
  int src, dst;
  if (e < EE) { src = ei[e]; dst = ei[EE + e]; } else { src = dst = e - EE; }

  const float4* xl = (const float4*)(XL + (size_t)src * HID) + lane * 2;
  const float4* xr = (const float4*)(XR + (size_t)dst * HID) + lane * 2;
  const float4* at = (const float4*)att + lane * 2;

  float p = 0.f;
#pragma unroll
  for (int q = 0; q < 2; ++q) {
    float4 a = xl[q], b = xr[q], w = at[q];
    float v;
    v = a.x + b.x; v = v > 0.f ? v : NEG_SLOPE * v; p += v * w.x;
    v = a.y + b.y; v = v > 0.f ? v : NEG_SLOPE * v; p += v * w.y;
    v = a.z + b.z; v = v > 0.f ? v : NEG_SLOPE * v; p += v * w.z;
    v = a.w + b.w; v = v > 0.f ? v : NEG_SLOPE * v; p += v * w.w;
  }
  // reduce across the 8 lanes of this head
  p += __shfl_xor(p, 1, 32);
  p += __shfl_xor(p, 2, 32);
  p += __shfl_xor(p, 4, 32);
  if ((lane & 7) == 0) {
    int h = lane >> 3;
    logit[(size_t)e * HEADS + h] = p;
    atomicMax(&menc[(size_t)dst * HEADS + h], fenc(p));
  }
}

// ---------------- edge pass 2: exp + scatter sum (in-place over logit) ------------
__global__ void k_edge_expsum(float* __restrict__ logit, const unsigned* __restrict__ menc,
                              float* __restrict__ ssum, const int* __restrict__ ei) {
  int idx = blockIdx.x * blockDim.x + threadIdx.x;
  if (idx >= EA * HEADS) return;
  int e = idx >> 2, h = idx & 3;
  int dst = (e < EE) ? ei[EE + e] : (e - EE);
  float m = fdec(menc[(size_t)dst * HEADS + h]);
  if (!isfinite(m)) m = 0.f;
  float ex = __expf(logit[idx] - m);
  logit[idx] = ex;
  atomAddF(&ssum[(size_t)dst * HEADS + h], ex);
}

// ---------------- edge pass 3: alpha * xl[src] scatter-add ----------------
__global__ void __launch_bounds__(256)
k_edge_agg(const float* __restrict__ ex, const float* __restrict__ ssum,
           const float* __restrict__ XL, const int* __restrict__ ei,
           float* __restrict__ agg) {
  int e = blockIdx.x * blockDim.y + threadIdx.y;
  if (e >= EA) return;
  int lane = threadIdx.x;
  int src, dst;
  if (e < EE) { src = ei[e]; dst = ei[EE + e]; } else { src = dst = e - EE; }
  int h = lane >> 3;
  float alpha = ex[(size_t)e * HEADS + h] / (ssum[(size_t)dst * HEADS + h] + 1e-16f);

  const float4* xl = (const float4*)(XL + (size_t)src * HID) + lane * 2;
  float* out = agg + (size_t)dst * HID + lane * 8;
#pragma unroll
  for (int q = 0; q < 2; ++q) {
    float4 a = xl[q];
    atomAddF(&out[q * 4 + 0], alpha * a.x);
    atomAddF(&out[q * 4 + 1], alpha * a.y);
    atomAddF(&out[q * 4 + 2], alpha * a.z);
    atomAddF(&out[q * 4 + 3], alpha * a.w);
  }
}

// ---------------- bias + relu + residual + LayerNorm (wave per node) ----------------
__global__ void __launch_bounds__(256)
k_post_ln(const float* __restrict__ agg, const float* __restrict__ bconv,
          const float* __restrict__ res, const float* __restrict__ gma,
          const float* __restrict__ bta, float* __restrict__ Hout,
          _Float16* __restrict__ Hf16) {
  int n = blockIdx.x * blockDim.y + threadIdx.y;
  if (n >= NN) return;
  int lane = threadIdx.x, c0 = lane * 8;
  size_t base = (size_t)n * HID + c0;

  float v[8];
#pragma unroll
  for (int j = 0; j < 8; ++j) {
    float t = agg[base + j] + bconv[c0 + j];
    v[j] = fmaxf(t, 0.f) + res[base + j];
  }
  float s1 = 0.f, s2 = 0.f;
#pragma unroll
  for (int j = 0; j < 8; ++j) { s1 += v[j]; s2 += v[j] * v[j]; }
#pragma unroll
  for (int off = 1; off < 32; off <<= 1) {
    s1 += __shfl_xor(s1, off, 32);
    s2 += __shfl_xor(s2, off, 32);
  }
  float mean = s1 * (1.f / HID);
  float var  = s2 * (1.f / HID) - mean * mean;
  float rstd = rsqrtf(var + 1e-5f);
#pragma unroll
  for (int j = 0; j < 8; ++j) {
    float o = (v[j] - mean) * rstd * gma[c0 + j] + bta[c0 + j];
    Hout[base + j] = o;
    Hf16[base + j] = (_Float16)o;
  }
}

// ---------------- global mean pool (scatter) ----------------
__global__ void __launch_bounds__(256)
k_pool(const float* __restrict__ H, const int* __restrict__ batch,
       float* __restrict__ sums, float* __restrict__ cnt) {
  int n = blockIdx.x * blockDim.y + threadIdx.y;
  if (n >= NN) return;
  int lane = threadIdx.x, c0 = lane * 8;
  int g = batch[n];
  const float* src = H + (size_t)n * HID + c0;
  float* dst = sums + (size_t)g * HID + c0;
#pragma unroll
  for (int j = 0; j < 8; ++j) atomAddF(&dst[j], src[j]);
  if (lane == 0) atomAddF(&cnt[g], 1.f);
}

// ---------------- classifier: pooled @ Wc + b_c ----------------
__global__ void k_cls(const float* __restrict__ sums, const float* __restrict__ cnt,
                      const float* __restrict__ Wc, const float* __restrict__ bc,
                      float* __restrict__ out) {
  int idx = blockIdx.x * blockDim.x + threadIdx.x;
  if (idx >= NGRAPH * NCLS) return;
  int g = idx / NCLS, c = idx % NCLS;
  float inv = 1.f / fmaxf(cnt[g], 1.f);
  float acc = bc[c];
  for (int k = 0; k < HID; ++k)
    acc += sums[(size_t)g * HID + k] * inv * Wc[(size_t)k * NCLS + c];
  out[idx] = acc;
}

// ---------------- host launcher ----------------
static inline int cdiv(long long a, long long b) { return (int)((a + b - 1) / b); }

extern "C" void kernel_launch(void* const* d_in, const int* in_sizes, int n_in,
                              void* d_out, int out_size, void* d_ws, size_t ws_size,
                              hipStream_t stream) {
  const float* x       = (const float*)d_in[0];
  const int*   ei      = (const int*)d_in[1];
  const int*   batch   = (const int*)d_in[2];
  const int*   deg     = (const int*)d_in[3];
  const float* Win     = (const float*)d_in[4];
  const float* b_in    = (const float*)d_in[5];
  const float* deg_emb = (const float*)d_in[6];
  const float* Wl      = (const float*)d_in[7];   // [L,HID,HID]
  const float* Wr      = (const float*)d_in[8];   // [L,HID,HID]
  const float* att     = (const float*)d_in[9];   // [L,HEADS,HD]
  const float* b_conv  = (const float*)d_in[10];  // [L,HID]
  const float* gma     = (const float*)d_in[11];  // [L,HID]
  const float* bta     = (const float*)d_in[12];  // [L,HID]
  const float* Wc      = (const float*)d_in[13];  // [HID,NC]
  const float* b_c     = (const float*)d_in[14];
  float* out = (float*)d_out;

  // ---- carve workspace ----
  char* w = (char*)d_ws;
  auto alloc = [&](size_t bytes) -> char* {
    char* p = w;
    w += (bytes + 255) & ~(size_t)255;
    return p;
  };
  _Float16* Xh   = (_Float16*)alloc((size_t)NN * INC * 2);
  float*    Ha   = (float*)alloc((size_t)NN * HID * 4);
  float*    Hb   = (float*)alloc((size_t)NN * HID * 4);
  _Float16* Hh   = (_Float16*)alloc((size_t)NN * HID * 2);
  float*    XL   = (float*)alloc((size_t)NN * HID * 4);
  float*    XR   = (float*)alloc((size_t)NN * HID * 4);   // reused as aggregation buffer
  float*    LG   = (float*)alloc((size_t)EA * HEADS * 4); // logits, then exp in-place
  unsigned* MENC = (unsigned*)alloc((size_t)NN * HEADS * 4);
  float*    SSUM = (float*)alloc((size_t)NN * HEADS * 4);
  float*    PSUM = (float*)alloc((size_t)NGRAPH * HID * 4);
  float*    PCNT = (float*)alloc((size_t)NGRAPH * 4);
  _Float16* PWin = (_Float16*)alloc((size_t)INC * HID * 2);
  _Float16* PWl[NLAYER];
  _Float16* PWr[NLAYER];
  for (int l = 0; l < NLAYER; ++l) PWl[l] = (_Float16*)alloc((size_t)HID * HID * 2);
  for (int l = 0; l < NLAYER; ++l) PWr[l] = (_Float16*)alloc((size_t)HID * HID * 2);

  dim3 blkWave(32, 8);  // 8 wave32 waves per block

  // ---- pack weights into WMMA B-fragment layout ----
  k_pack_w<<<cdiv(INC * HID, 256), 256, 0, stream>>>(Win, PWin, INC, HID);
  for (int l = 0; l < NLAYER; ++l) {
    k_pack_w<<<cdiv(HID * HID, 256), 256, 0, stream>>>(Wl + (size_t)l * HID * HID, PWl[l], HID, HID);
    k_pack_w<<<cdiv(HID * HID, 256), 256, 0, stream>>>(Wr + (size_t)l * HID * HID, PWr[l], HID, HID);
  }

  // ---- input projection: h = x@Win + b_in + deg_emb[deg] ----
  k_cast_f16<<<cdiv((size_t)NN * INC, 256), 256, 0, stream>>>(x, Xh, NN * INC);
  {
    int strips = (NN / 16) * (HID / 64);
    k_wmma_gemm4<INC / 32><<<cdiv(strips, 8), blkWave, 0, stream>>>(
        Xh, PWin, b_in, deg_emb, deg, Ha, Hh, NN, HID);
  }

  float* hbuf[2] = { Ha, Hb };
  for (int l = 0; l < NLAYER; ++l) {
    float* cur = hbuf[l & 1];
    float* nxt = hbuf[(l + 1) & 1];
    int strips = (NN / 16) * (HID / 64);

    // xl = h@Wl, xr = h@Wr  (WMMA)
    k_wmma_gemm4<HID / 32><<<cdiv(strips, 8), blkWave, 0, stream>>>(
        Hh, PWl[l], nullptr, nullptr, nullptr, XL, nullptr, NN, HID);
    k_wmma_gemm4<HID / 32><<<cdiv(strips, 8), blkWave, 0, stream>>>(
        Hh, PWr[l], nullptr, nullptr, nullptr, XR, nullptr, NN, HID);

    // scatter-softmax over incoming edges
    hipMemsetAsync(MENC, 0, (size_t)NN * HEADS * 4, stream);   // 0 == encoded -max
    hipMemsetAsync(SSUM, 0, (size_t)NN * HEADS * 4, stream);
    k_edge_logits<<<cdiv(EA, 8), blkWave, 0, stream>>>(
        XL, XR, ei, att + (size_t)l * HEADS * HD, LG, MENC);
    hipMemsetAsync(XR, 0, (size_t)NN * HID * 4, stream);       // XR now = aggregation target
    k_edge_expsum<<<cdiv((size_t)EA * HEADS, 256), 256, 0, stream>>>(LG, MENC, SSUM, ei);
    k_edge_agg<<<cdiv(EA, 8), blkWave, 0, stream>>>(LG, SSUM, XL, ei, XR);

    // bias + relu + residual + layernorm; produce f32 h and f16 copy for next GEMMs
    k_post_ln<<<cdiv(NN, 8), blkWave, 0, stream>>>(
        XR, b_conv + (size_t)l * HID, cur, gma + (size_t)l * HID, bta + (size_t)l * HID,
        nxt, Hh);
  }

  // ---- global mean pool + classifier ----
  float* hfin = hbuf[NLAYER & 1];
  hipMemsetAsync(PSUM, 0, (size_t)NGRAPH * HID * 4, stream);
  hipMemsetAsync(PCNT, 0, (size_t)NGRAPH * 4, stream);
  k_pool<<<cdiv(NN, 8), blkWave, 0, stream>>>(hfin, batch, PSUM, PCNT);
  k_cls<<<cdiv(NGRAPH * NCLS, 128), 128, 0, stream>>>(PSUM, PCNT, Wc, b_c, out);
}
